// RoboticManipulationGNN_36971078484409
// MI455X (gfx1250) — matrix operand order repaired and checked by reading, not compiled, not run on hardware
//
#include <hip/hip_runtime.h>

typedef __attribute__((ext_vector_type(16))) _Float16 v16h;
typedef __attribute__((ext_vector_type(8)))  float    v8f;

union Frag16 { uint4 q[2]; v16h v; };

#define N_NODES 100000
#define N_EDGES 1600000
#define IN_DIM  128
#define HID_DIM 256

// ---------------- elementwise / prep kernels ----------------

__global__ void k_f32_to_f16(const float* __restrict__ x, _Float16* __restrict__ y, int n) {
  int i = blockIdx.x * blockDim.x + threadIdx.x;
  if (i < n) y[i] = (_Float16)x[i];
}

// Wt[n*K + k] = W[k*N + n]  (transpose + f16 convert), W is [K,N]
__global__ void k_w_transpose(const float* __restrict__ W, _Float16* __restrict__ Wt, int K, int N) {
  int i = blockIdx.x * blockDim.x + threadIdx.x;
  if (i < K * N) {
    int k = i / N, n = i % N;
    Wt[n * K + k] = (_Float16)W[i];
  }
}

__global__ void k_deg_init(float* __restrict__ deg, int n) {
  int i = blockIdx.x * blockDim.x + threadIdx.x;
  if (i < n) deg[i] = 1.0f;   // self loop
}

__global__ void k_deg_count(const long long* __restrict__ ei, float* __restrict__ deg, int E) {
  int e = blockIdx.x * blockDim.x + threadIdx.x;
  if (e < E) atomicAdd(&deg[(int)ei[(size_t)E + e]], 1.0f);
}

__global__ void k_rsqrt_inplace(float* __restrict__ deg, int n) {
  int i = blockIdx.x * blockDim.x + threadIdx.x;
  if (i < n) deg[i] = rsqrtf(deg[i]);   // deg >= 1 always
}

__global__ void k_norm(const long long* __restrict__ ei, const float* __restrict__ dinv,
                       float* __restrict__ norm, int E) {
  int e = blockIdx.x * blockDim.x + threadIdx.x;
  if (e < E) norm[e] = dinv[(int)ei[e]] * dinv[(int)ei[(size_t)E + e]];
}

// ---------------- WMMA GEMM:  C[M,N=256] = A[M,K] * W[K,256]  ----------------
// A: f16 row-major [M,K]; Wt: f16 [N,K] (W transposed); C: f32 [M,N]
// Block: 256 threads = 8 waves; each wave computes a 16-row strip x full N.
// Wt is staged into LDS with CDNA5 async global->LDS copies (ASYNCcnt).
__global__ void __launch_bounds__(256)
k_gemm_wmma_f16(const _Float16* __restrict__ A, const _Float16* __restrict__ Wt,
                float* __restrict__ C, int M, int K) {
  const int N = 256;
  extern __shared__ _Float16 sWt[];          // N*K f16
  const int tid = threadIdx.x;

  // ---- async stage Wt -> LDS (each lane copies 16B per iteration) ----
  {
    const int iters = (N * K * 2) / (256 * 16);
    unsigned ldsBase = (unsigned)(size_t)(void*)sWt;      // low 32 bits = LDS byte addr
    unsigned long long g = (unsigned long long)(const char*)Wt + (unsigned long long)tid * 16u;
    unsigned l = ldsBase + (unsigned)tid * 16u;
    for (int i = 0; i < iters; ++i) {
      unsigned laddr = l + (unsigned)i * 4096u;
      unsigned long long gaddr = g + (unsigned long long)i * 4096u;
      asm volatile("global_load_async_to_lds_b128 %0, %1, off"
                   :: "v"(laddr), "v"(gaddr) : "memory");
    }
    asm volatile("s_wait_asynccnt 0" ::: "memory");
  }
  __syncthreads();

  const int wave = tid >> 5;
  const int lane = tid & 31;
  const int half = lane >> 4;    // 0 or 1
  const int l16  = lane & 15;

  const int rowBase = blockIdx.x * 128 + wave * 16;
  int arow = rowBase + l16;                  // A-row this lane loads
  if (arow > M - 1) arow = M - 1;            // clamp (stores are guarded)

  v8f acc[16];
  #pragma unroll
  for (int t = 0; t < 16; ++t)
    #pragma unroll
    for (int r = 0; r < 8; ++r) acc[t][r] = 0.0f;

  for (int k0 = 0; k0 < K; k0 += 32) {
    // A fragment 16x32: VGPR0-3 = halfs [half*8, half*8+8), VGPR4-7 = +16
    Frag16 a;
    const _Float16* ap = A + (size_t)arow * K + k0 + half * 8;
    a.q[0] = *(const uint4*)(ap);
    a.q[1] = *(const uint4*)(ap + 16);

    #pragma unroll
    for (int nt = 0; nt < 16; ++nt) {
      // B fragment 32x16: lane column n = nt*16+l16, K halfs [half*8, half*8+8) then +16
      Frag16 b;
      const _Float16* bp = sWt + (size_t)(nt * 16 + l16) * K + k0 + half * 8;
      b.q[0] = *(const uint4*)(bp);
      b.q[1] = *(const uint4*)(bp + 16);
      acc[nt] = __builtin_amdgcn_wmma_f32_16x16x32_f16(
          false, a.v, false, b.v, (short)0, acc[nt], false, false);
    }
  }

  // C layout: VGPR r -> tile row half*8 + r, column l16
  #pragma unroll
  for (int nt = 0; nt < 16; ++nt) {
    #pragma unroll
    for (int r = 0; r < 8; ++r) {
      int m = rowBase + half * 8 + r;
      if (m < M) C[(size_t)m * N + nt * 16 + l16] = acc[nt][r];
    }
  }
}

// ---------------- aggregation ----------------

// agg[i] = dinv[node]^2 * h[i]   (self loop contribution, also zero-inits agg)
__global__ void k_selfloop_init(const float* __restrict__ dinv, const float* __restrict__ h,
                                float* __restrict__ agg, int total) {
  int i = blockIdx.x * blockDim.x + threadIdx.x;
  if (i < total) {
    float di = dinv[i >> 8];
    agg[i] = di * di * h[i];
  }
}

// one edge per 64 lanes; each lane handles 4 consecutive features (float4 gather)
__global__ void k_edge_scatter(const long long* __restrict__ ei, const float* __restrict__ norm,
                               const float* __restrict__ h, float* __restrict__ agg, int E) {
  unsigned t = blockIdx.x * blockDim.x + threadIdx.x;
  int e = (int)(t >> 6);
  if (e >= E) return;
  int f = (int)(t & 63u) << 2;
  int s = (int)ei[e];
  int d = (int)ei[(size_t)E + e];
  float w = norm[e];
  float4 hv = *(const float4*)(h + (size_t)s * 256 + f);
  float* dst = agg + (size_t)d * 256 + f;
  atomicAdd(dst + 0, w * hv.x);
  atomicAdd(dst + 1, w * hv.y);
  atomicAdd(dst + 2, w * hv.z);
  atomicAdd(dst + 3, w * hv.w);
}

// z[i] = f16(relu(agg[i] + b[i%256]))
__global__ void k_bias_relu_f16(const float* __restrict__ agg, const float* __restrict__ b,
                                _Float16* __restrict__ z, int total) {
  int i = blockIdx.x * blockDim.x + threadIdx.x;
  if (i < total) {
    float v = agg[i] + b[i & 255];
    z[i] = (_Float16)fmaxf(v, 0.0f);
  }
}

__global__ void k_zero256(float* __restrict__ p) { p[threadIdx.x] = 0.0f; }

// pooled[f] += sum over node chunk of relu(agg[n*256+f] + b[f])
__global__ void k_pool(const float* __restrict__ agg, const float* __restrict__ b,
                       float* __restrict__ pooled, int nnodes) {
  int f = threadIdx.x;
  int chunk = (nnodes + gridDim.x - 1) / gridDim.x;
  int n0 = blockIdx.x * chunk;
  int n1 = n0 + chunk; if (n1 > nnodes) n1 = nnodes;
  float bf = b[f];
  float s = 0.0f;
  for (int n = n0; n < n1; ++n) s += fmaxf(agg[(size_t)n * 256 + f] + bf, 0.0f);
  atomicAdd(&pooled[f], s);
}

__global__ void k_final(const float* __restrict__ pooled, const float* __restrict__ Wout,
                        const float* __restrict__ bout, float* __restrict__ out, int nnodes) {
  __shared__ float red[256];
  int t = threadIdx.x;
  red[t] = pooled[t] * Wout[t];
  __syncthreads();
  for (int s = 128; s > 0; s >>= 1) {
    if (t < s) red[t] += red[t + s];
    __syncthreads();
  }
  if (t == 0) out[0] = red[0] / (float)nnodes + bout[0];
}

// ---------------- launch ----------------

extern "C" void kernel_launch(void* const* d_in, const int* in_sizes, int n_in,
                              void* d_out, int out_size, void* d_ws, size_t ws_size,
                              hipStream_t stream) {
  const float*      x    = (const float*)d_in[0];
  const long long*  ei   = (const long long*)d_in[1];   // int64 [2,E]
  const float*      W1   = (const float*)d_in[2];
  const float*      b1   = (const float*)d_in[3];
  const float*      W2   = (const float*)d_in[4];
  const float*      b2   = (const float*)d_in[5];
  const float*      Wout = (const float*)d_in[6];
  const float*      bout = (const float*)d_in[7];
  float*            out  = (float*)d_out;

  const int Nn = in_sizes[0] / IN_DIM;    // 100000
  const int E  = in_sizes[1] / 2;         // 1600000

  // workspace layout (4KB aligned)
  char* ws = (char*)d_ws;
  float*    deg  = (float*)(ws + 0);              // N*4          (becomes dinv)
  float*    nrm  = (float*)(ws + 401408);         // E*4
  float*    pool = (float*)(ws + 6803456);        // 256*4
  _Float16* w1t  = (_Float16*)(ws + 6807552);     // 256*128*2
  _Float16* w2t  = (_Float16*)(ws + 6873088);     // 256*256*2
  _Float16* xb   = (_Float16*)(ws + 7004160);     // N*128*2
  _Float16* zb   = (_Float16*)(ws + 32604160);    // N*256*2
  float*    h    = (float*)(ws + 83804160);       // N*256*4
  float*    agg  = (float*)(ws + 186204160);      // N*256*4

  const int B = 256;
  const int totX = Nn * IN_DIM;       // 12.8M
  const int totH = Nn * HID_DIM;      // 25.6M

  // 1) precision conversion + weight transposes
  k_f32_to_f16<<<(totX + B - 1) / B, B, 0, stream>>>(x, xb, totX);
  k_w_transpose<<<(IN_DIM * HID_DIM + B - 1) / B, B, 0, stream>>>(W1, w1t, IN_DIM, HID_DIM);
  k_w_transpose<<<(HID_DIM * HID_DIM + B - 1) / B, B, 0, stream>>>(W2, w2t, HID_DIM, HID_DIM);

  // 2) degrees + symmetric norm
  k_deg_init<<<(Nn + B - 1) / B, B, 0, stream>>>(deg, Nn);
  k_deg_count<<<(E + B - 1) / B, B, 0, stream>>>(ei, deg, E);
  k_rsqrt_inplace<<<(Nn + B - 1) / B, B, 0, stream>>>(deg, Nn);
  k_norm<<<(E + B - 1) / B, B, 0, stream>>>(ei, deg, nrm, E);

  // 3) layer 1: h = x @ W1  (WMMA), aggregate, bias+relu -> f16
  {
    dim3 grid((Nn + 127) / 128);
    size_t lds = (size_t)HID_DIM * IN_DIM * 2;   // 64KB
    k_gemm_wmma_f16<<<grid, B, lds, stream>>>(xb, w1t, h, Nn, IN_DIM);
  }
  k_selfloop_init<<<(totH + B - 1) / B, B, 0, stream>>>(deg, h, agg, totH);
  k_edge_scatter<<<(E * 64u + B - 1) / B, B, 0, stream>>>(ei, nrm, h, agg, E);
  k_bias_relu_f16<<<(totH + B - 1) / B, B, 0, stream>>>(agg, b1, zb, totH);

  // 4) layer 2: h = z1 @ W2  (WMMA), aggregate
  {
    dim3 grid((Nn + 127) / 128);
    size_t lds = (size_t)HID_DIM * HID_DIM * 2;  // 128KB
    k_gemm_wmma_f16<<<grid, B, lds, stream>>>(zb, w2t, h, Nn, HID_DIM);
  }
  k_selfloop_init<<<(totH + B - 1) / B, B, 0, stream>>>(deg, h, agg, totH);
  k_edge_scatter<<<(E * 64u + B - 1) / B, B, 0, stream>>>(ei, nrm, h, agg, E);

  // 5) mean pool (bias+relu fused) + output projection
  k_zero256<<<1, 256, 0, stream>>>(pool);
  k_pool<<<1024, 256, 0, stream>>>(agg, b2, pool, Nn);
  k_final<<<1, 256, 0, stream>>>(pool, Wout, bout, out, Nn);
}
